// DirectedGNNEncoder_63445256896874
// MI455X (gfx1250) — matrix operand order repaired and checked by reading, not compiled, not run on hardware
//
#include <hip/hip_runtime.h>
#include <hip/hip_bf16.h>

#define DD   256   // feature dim (= H*C)
#define HH   8     // heads
#define CC   32    // channels per head
#define NEGS 0.2f  // leaky-relu slope

typedef __attribute__((ext_vector_type(16))) __bf16 v16bf;
typedef __attribute__((ext_vector_type(8)))  float  v8f;

__device__ __forceinline__ void atomicMaxF(float* addr, float val) {
  // IEEE ordering trick: positive floats compare as ints, negative as reversed uints.
  if (val >= 0.0f) atomicMax((int*)addr, __float_as_int(val));
  else             atomicMin((unsigned int*)addr, __float_as_uint(val));
}

// ---------------------------------------------------------------------------
// One-time: transpose all 8 weight matrices (Wl[0..3], Wr[0..3]; [K=256,M=256]
// row-major f32) into column-major bf16 WT[mat][m][k] so B fragments are
// contiguous 32B per lane.
// ---------------------------------------------------------------------------
__global__ void prep_weights_kernel(const float* __restrict__ Wl,
                                    const float* __restrict__ Wr,
                                    unsigned short* __restrict__ WT) {
  int tid = blockIdx.x * blockDim.x + threadIdx.x;   // [0, 8*65536)
  int mat = tid >> 16;
  int m   = (tid >> 8) & 255;
  int k   = tid & 255;
  const float* Wsrc = (mat < 4) ? (Wl + (size_t)mat * 65536)
                                : (Wr + (size_t)(mat - 4) * 65536);
  __bf16 b = (__bf16)Wsrc[k * 256 + m];
  WT[tid] = __builtin_bit_cast(unsigned short, b);
}

// ---------------------------------------------------------------------------
// Out[n, m] = bf16(X[n,:]) @ W[:, m] + bias[m]   via v_wmma_f32_16x16x32_bf16
// Block = 256 threads = 8 waves. Wave w: row tile rt = blockIdx.x*8+w,
// columns [blockIdx.y*128, +128) as 8 16x16 tiles. K loop: 8 steps of 32.
// Per K-step: issue all A + B loads up front (one wait covers the group),
// then run the 8-WMMA chain back-to-back.
// ---------------------------------------------------------------------------
__global__ void gemm_wmma_kernel(const float* __restrict__ X,
                                 const unsigned short* __restrict__ WTu,
                                 const float* __restrict__ bias,
                                 float* __restrict__ Out,
                                 int nrows) {
  const __bf16* WT = reinterpret_cast<const __bf16*>(WTu);
  const int wave = threadIdx.x >> 5;
  const int lane = threadIdx.x & 31;
  const int rt   = blockIdx.x * 8 + wave;
  const int row0 = rt * 16;
  if (row0 >= nrows) return;

  const int r  = lane & 15;       // A row within tile
  const int hi = lane >> 4;       // lane half
  const int col0 = blockIdx.y * 128;

  const bool rowOk = (row0 + r) < nrows;   // always true when nrows % 16 == 0
  const float* arow = X + (size_t)(row0 + r) * DD;
  const int akoff = hi * 16;      // A: lanes 0-15 -> K 0..15; 16-31 -> K 16..31
  const int bn    = lane & 15;    // B column within tile
  const int bkoff = hi * 16;      // B: lanes 0-15 -> K 0..15; 16-31 -> K 16..31

  // Per-lane B base pointers (column-major bf16, contiguous K)
  const __bf16* bbase[8];
#pragma unroll
  for (int t = 0; t < 8; ++t)
    bbase[t] = WT + (size_t)(col0 + t * 16 + bn) * DD + bkoff;

  v8f acc[8];
#pragma unroll
  for (int t = 0; t < 8; ++t) {
    v8f z = {0.f, 0.f, 0.f, 0.f, 0.f, 0.f, 0.f, 0.f};
    acc[t] = z;
  }

  for (int k0 = 0; k0 < DD; k0 += 32) {
    // ---- issue all loads for this K-step up front ----
    float4 a0, a1, a2, a3;
    if (rowOk) {
      const float* ap = arow + k0 + akoff;
      a0 = *(const float4*)(ap + 0);
      a1 = *(const float4*)(ap + 4);
      a2 = *(const float4*)(ap + 8);
      a3 = *(const float4*)(ap + 12);
    } else {
      a0 = a1 = a2 = a3 = make_float4(0.f, 0.f, 0.f, 0.f);
    }
    v16bf b[8];
#pragma unroll
    for (int t = 0; t < 8; ++t)
      b[t] = *(const v16bf*)(bbase[t] + k0);

    // ---- convert A (overlaps with B loads in flight) ----
    v16bf a;
    a[0]  = (__bf16)a0.x; a[1]  = (__bf16)a0.y; a[2]  = (__bf16)a0.z; a[3]  = (__bf16)a0.w;
    a[4]  = (__bf16)a1.x; a[5]  = (__bf16)a1.y; a[6]  = (__bf16)a1.z; a[7]  = (__bf16)a1.w;
    a[8]  = (__bf16)a2.x; a[9]  = (__bf16)a2.y; a[10] = (__bf16)a2.z; a[11] = (__bf16)a2.w;
    a[12] = (__bf16)a3.x; a[13] = (__bf16)a3.y; a[14] = (__bf16)a3.z; a[15] = (__bf16)a3.w;

    // ---- 8 WMMAs back-to-back ----
#pragma unroll
    for (int t = 0; t < 8; ++t) {
      acc[t] = __builtin_amdgcn_wmma_f32_16x16x32_bf16(
          /*neg_a=*/false, a, /*neg_b=*/false, b[t],
          /*c_mod=*/(short)0, acc[t], /*reuse_a=*/false, /*reuse_b=*/false);
    }
  }

  // C/D layout: lane half hi -> rows hi*8+v, column = lane&15 within tile.
  const int rowadd = hi * 8;
#pragma unroll
  for (int t = 0; t < 8; ++t) {
    const int col = col0 + t * 16 + bn;
    const float bv = bias[col];
#pragma unroll
    for (int v = 0; v < 8; ++v) {
      const int rr = row0 + rowadd + v;
      if (rr < nrows) Out[(size_t)rr * DD + col] = acc[t][v] + bv;
    }
  }
}

// ---------------------------------------------------------------------------
// Init per-node state: out = bias (empty segments end as pure bias),
// segmax = -inf, segsum = 0.
// ---------------------------------------------------------------------------
__global__ void init_nodes_kernel(float* __restrict__ segmax,
                                  float* __restrict__ segsum,
                                  float* __restrict__ out,
                                  const float* __restrict__ bias, int N) {
  int tid = blockIdx.x * blockDim.x + threadIdx.x;
  if (tid >= N * DD) return;
  int c = tid & 255;
  out[tid] = bias[c];
  if (c < HH) {
    int n = tid >> 8;
    segmax[n * HH + c] = -INFINITY;
    segsum[n * HH + c] = 0.0f;
  }
}

// ---------------------------------------------------------------------------
// logits[e,h] = sum_c leakyrelu(hl[src,h,c]+hr[dst,h,c]) * att[h,c];
// scatter-max into segmax[dst,h].
// ---------------------------------------------------------------------------
__global__ void edge_logits_kernel(const float* __restrict__ hl,
                                   const float* __restrict__ hr,
                                   const float* __restrict__ att,
                                   const int* __restrict__ src,
                                   const int* __restrict__ dst,
                                   float* __restrict__ P,
                                   float* __restrict__ segmax, int E_) {
  int tid = blockIdx.x * blockDim.x + threadIdx.x;
  if (tid >= E_ * HH) return;
  int e = tid >> 3, h = tid & 7;
  int sN = src[e], dN = dst[e];
  const float4* a4 = (const float4*)(hl + (size_t)sN * DD + h * CC);
  const float4* b4 = (const float4*)(hr + (size_t)dN * DD + h * CC);
  const float4* w4 = (const float4*)(att + h * CC);
  float acc = 0.0f;
#pragma unroll
  for (int i = 0; i < CC / 4; ++i) {
    float4 x = a4[i], y = b4[i], w = w4[i];
    float z0 = x.x + y.x, z1 = x.y + y.y, z2 = x.z + y.z, z3 = x.w + y.w;
    z0 = z0 > 0.f ? z0 : NEGS * z0;
    z1 = z1 > 0.f ? z1 : NEGS * z1;
    z2 = z2 > 0.f ? z2 : NEGS * z2;
    z3 = z3 > 0.f ? z3 : NEGS * z3;
    acc += z0 * w.x + z1 * w.y + z2 * w.z + z3 * w.w;
  }
  P[tid] = acc;
  atomicMaxF(segmax + (size_t)dN * HH + h, acc);
}

// p = exp(logit - segmax[dst,h]); scatter-add into segsum[dst,h].
__global__ void edge_exp_kernel(float* __restrict__ P,
                                const float* __restrict__ segmax,
                                float* __restrict__ segsum,
                                const int* __restrict__ dst, int E_) {
  int tid = blockIdx.x * blockDim.x + threadIdx.x;
  if (tid >= E_ * HH) return;
  int e = tid >> 3, h = tid & 7;
  int dN = dst[e];
  float m = segmax[(size_t)dN * HH + h];
  float p = __expf(P[tid] - m);
  P[tid] = p;
  atomicAdd(segsum + (size_t)dN * HH + h, p);
}

// out[dst,h,c] += (p / (segsum+eps)) * hl[src,h,c]   (4 channels per thread)
__global__ void edge_scatter_kernel(const float* __restrict__ P,
                                    const float* __restrict__ segsum,
                                    const float* __restrict__ hl,
                                    const int* __restrict__ src,
                                    const int* __restrict__ dst,
                                    float* __restrict__ out, int E_) {
  int tid = blockIdx.x * blockDim.x + threadIdx.x;
  if (tid >= E_ * 64) return;
  int e = tid >> 6;
  int q = tid & 63;
  int h = q >> 3, c4 = q & 7;
  int sN = src[e], dN = dst[e];
  float denom = segsum[(size_t)dN * HH + h] + 1e-16f;
  float alpha = P[(size_t)e * HH + h] / denom;
  const float4 v = *(const float4*)(hl + (size_t)sN * DD + h * CC + c4 * 4);
  float* o = out + (size_t)dN * DD + h * CC + c4 * 4;
  atomicAdd(o + 0, alpha * v.x);
  atomicAdd(o + 1, alpha * v.y);
  atomicAdd(o + 2, alpha * v.z);
  atomicAdd(o + 3, alpha * v.w);
}

__global__ void relu_kernel(float* __restrict__ x, int n) {
  int tid = blockIdx.x * blockDim.x + threadIdx.x;
  if (tid < n) x[tid] = fmaxf(x[tid], 0.0f);
}

// ---------------------------------------------------------------------------
extern "C" void kernel_launch(void* const* d_in, const int* in_sizes, int n_in,
                              void* d_out, int out_size, void* d_ws, size_t ws_size,
                              hipStream_t stream) {
  const float* s    = (const float*)d_in[0];
  const float* t    = (const float*)d_in[1];
  const float* Wl   = (const float*)d_in[2];
  const float* bl   = (const float*)d_in[3];
  const float* Wr   = (const float*)d_in[4];
  const float* br   = (const float*)d_in[5];
  const float* att  = (const float*)d_in[6];
  const float* bias = (const float*)d_in[7];
  const int*   ei   = (const int*)d_in[8];

  const int Nn = in_sizes[0] / DD;   // 30000
  const int Ee = in_sizes[8] / 2;    // 480000
  const int* srcI = ei;
  const int* dstI = ei + Ee;

  // ---- workspace carve-up (256B aligned) ----
  char* wsp = (char*)d_ws;
  auto carve = [&](size_t bytes) -> char* {
    char* p = wsp;
    wsp += (bytes + 255) & ~(size_t)255;
    return p;
  };
  unsigned short* WT = (unsigned short*)carve((size_t)8 * 65536 * sizeof(unsigned short));
  float* HL   = (float*)carve((size_t)Nn * DD * sizeof(float));
  float* HR   = (float*)carve((size_t)Nn * DD * sizeof(float));
  float* P    = (float*)carve((size_t)Ee * HH * sizeof(float));
  float* SMAX = (float*)carve((size_t)Nn * HH * sizeof(float));
  float* SSUM = (float*)carve((size_t)Nn * HH * sizeof(float));
  float* XS0  = (float*)carve((size_t)Nn * DD * sizeof(float));
  float* XT0  = (float*)carve((size_t)Nn * DD * sizeof(float));

  float* outSfin = (float*)d_out;
  float* outTfin = outSfin + (size_t)Nn * DD;

  // ---- weight prep (8 * 256 * 256 elements) ----
  prep_weights_kernel<<<(8 * 65536) / 256, 256, 0, stream>>>(Wl, Wr, WT);

  const int rowTiles   = (Nn + 15) / 16;
  const dim3 gemmGrid((rowTiles + 7) / 8, 2);
  const int nodeBlocks = (Nn * DD + 255) / 256;
  const int ehBlocks   = (Ee * HH + 255) / 256;
  const int scBlocks   = (Ee * 64 + 255) / 256;

  auto run_gatv2 = [&](const float* xsrc, const float* xdst,
                       const int* sArr, const int* dArr,
                       int enc, float* outBuf) {
    const unsigned short* WlT = WT + (size_t)enc * 65536;
    const unsigned short* WrT = WT + (size_t)(4 + enc) * 65536;
    gemm_wmma_kernel<<<gemmGrid, 256, 0, stream>>>(xsrc, WlT, bl + enc * DD, HL, Nn);
    gemm_wmma_kernel<<<gemmGrid, 256, 0, stream>>>(xdst, WrT, br + enc * DD, HR, Nn);
    init_nodes_kernel<<<nodeBlocks, 256, 0, stream>>>(SMAX, SSUM, outBuf, bias + enc * DD, Nn);
    edge_logits_kernel<<<ehBlocks, 256, 0, stream>>>(HL, HR, att + enc * DD, sArr, dArr, P, SMAX, Ee);
    edge_exp_kernel<<<ehBlocks, 256, 0, stream>>>(P, SMAX, SSUM, dArr, Ee);
    edge_scatter_kernel<<<scBlocks, 256, 0, stream>>>(P, SSUM, HL, sArr, dArr, outBuf, Ee);
  };

  const float* xs = s;
  const float* xt = t;
  for (int l = 0; l < 2; ++l) {
    float* oS = (l == 0) ? XS0 : outSfin;
    float* oT = (l == 0) ? XT0 : outTfin;
    // source encoder: messages src -> dst
    run_gatv2(xs, xt, srcI, dstI, 2 * l + 0, oS);
    // target encoder: flipped edges dst -> src
    run_gatv2(xt, xs, dstI, srcI, 2 * l + 1, oT);
    if (l == 0) {
      relu_kernel<<<nodeBlocks, 256, 0, stream>>>(XS0, Nn * DD);
      relu_kernel<<<nodeBlocks, 256, 0, stream>>>(XT0, Nn * DD);
      xs = XS0;
      xt = XT0;
    }
  }
}
